// Attention_64037962383614
// MI455X (gfx1250) — matrix operand order repaired and checked by reading, not compiled, not run on hardware
//
#include <hip/hip_runtime.h>
#include <hip/hip_fp16.h>

// ---------------------------------------------------------------------------
// MHA forward for MI455X (gfx1250), wave32, WMMA f32_16x16x32_f16.
//   x:(4,2048,1024) f32; W_q/k/v/o:(1024,1024) f32 [out,in]; b_o:(1024,)
//   out = softmax(QK^T/8) V  projected, (4,2048,1024) f32
// V is produced transposed ([b][head][d][l]) so every WMMA B-fragment in the
// whole pipeline is a contiguous 2x ds_load_b128.
// ---------------------------------------------------------------------------

typedef __attribute__((ext_vector_type(16))) _Float16 v16h;
typedef __attribute__((ext_vector_type(8)))  float    v8f;
typedef __attribute__((ext_vector_type(4)))  _Float16 h4;

#define WMMA_F16(A, B, C) \
    __builtin_amdgcn_wmma_f32_16x16x32_f16(false, (A), false, (B), (short)0, (C), false, false)

// A-fragment (16x32 f16, MxK) from row-major f16 tile.
// k order used here (contiguous, same as frag_bT): lanes 0-15 -> M=lane, K=0..15; lanes 16-31 -> K=16..31.
static __device__ inline v16h frag_a(const _Float16* base, int stride, int lane) {
    const _Float16* row = base + (size_t)(lane & 15) * stride + ((lane & 16) ? 16 : 0);
    v16h a;
#pragma unroll
    for (int p = 0; p < 8; ++p) {
        int k = p << 1;   // 0,2,4,...,14
        a[2 * p]     = row[k];
        a[2 * p + 1] = row[k + 1];
    }
    return a;
}

// B-fragment (32x16 f16, KxN) from a tile laid out [n][k]
// (W[out][in], K[j][d], V^T[d][j]).  Contiguous: lowers to 2x ds_load_b128.
// k order used here (contiguous, same as frag_a): lane -> N=lane&15; lanes 0-15 K=0..15, lanes 16-31 K=16..31.
static __device__ inline v16h frag_bT(const _Float16* base, int stride, int lane) {
    const _Float16* p = base + (size_t)(lane & 15) * stride + ((lane & 16) ? 16 : 0);
    v16h b;
#pragma unroll
    for (int h = 0; h < 16; ++h) b[h] = p[h];
    return b;
}

// CDNA5 async memory->LDS copy (GVS form), tracked by ASYNCcnt.
// lds_addr: LDS byte address (flat-aperture low 32 bits == LDS offset per ISA
// aperture rules); goff: byte offset against the uniform 64-bit base in SGPRs.
static __device__ inline void async_load_b128(unsigned lds_addr, unsigned goff,
                                              const void* base) {
    asm volatile("global_load_async_to_lds_b128 %0, %1, %2"
                 :: "v"(lds_addr), "v"(goff), "s"(base)
                 : "memory");
}
static __device__ inline void wait_asynccnt0() {
    asm volatile("s_wait_asynccnt 0" ::: "memory");
}

// ---------------------------------------------------------------------------
// Kernel 1: Q/K/V projection.  Y = X @ W^T, f32 -> f16.
//   Q, K written as [b][head][l][64];  V written transposed as [b][head][64][l].
// Both layouts flatten to  idx = bh*131072 + l*sL + d*sD  with uniform strides,
// so the epilogue is branch-free.
// Block: 8 waves, 256x64 output tile; each wave owns a 32x64 strip
// (2 A-frags reuse 4 B-frags -> 8 WMMAs per 32-deep K chunk).  z selects W.
// ---------------------------------------------------------------------------
__global__ __launch_bounds__(256) void qkv_proj_kernel(
    const float* __restrict__ X,  const float* __restrict__ Wq,
    const float* __restrict__ Wk, const float* __restrict__ Wv,
    _Float16* __restrict__ Qh, _Float16* __restrict__ Kh, _Float16* __restrict__ Vt)
{
    __shared__ _Float16 lX[256][40];   // 80B rows (16B-aligned)
    __shared__ _Float16 lW[64][40];

    const int tid  = threadIdx.x;
    const int lane = tid & 31;
    const int w    = tid >> 5;
    const int r0   = blockIdx.x * 256;
    const int n0   = blockIdx.y * 64;
    const int z    = blockIdx.z;
    const float* Wm  = (z == 0) ? Wq : (z == 1) ? Wk : Wv;
    _Float16*    Out = (z == 0) ? Qh : (z == 1) ? Kh : Vt;
    // Uniform layout strides: Q/K row-major [l][d]; V transposed [d][l].
    const int sL = (z == 2) ? 1    : 64;
    const int sD = (z == 2) ? 2048 : 1;

    const v8f z8 = {0.f, 0.f, 0.f, 0.f, 0.f, 0.f, 0.f, 0.f};
    v8f acc[2][4] = {{z8, z8, z8, z8}, {z8, z8, z8, z8}};

    for (int kk = 0; kk < 1024; kk += 32) {
        __syncthreads();
        // X tile 256x32 f32 -> f16 (8 float4 per thread)
#pragma unroll
        for (int p = 0; p < 8; ++p) {
            int row = p * 32 + (tid >> 3);
            int col = (tid & 7) * 4;
            float4 v = *(const float4*)(X + (size_t)(r0 + row) * 1024 + kk + col);
            h4 hv = {(_Float16)v.x, (_Float16)v.y, (_Float16)v.z, (_Float16)v.w};
            *(h4*)&lX[row][col] = hv;
        }
        // W tile 64x32 f32 -> f16, stored [n][k]
#pragma unroll
        for (int p = 0; p < 2; ++p) {
            int row = p * 32 + (tid >> 3);
            int col = (tid & 7) * 4;
            float4 v = *(const float4*)(Wm + (size_t)(n0 + row) * 1024 + kk + col);
            h4 hv = {(_Float16)v.x, (_Float16)v.y, (_Float16)v.z, (_Float16)v.w};
            *(h4*)&lW[row][col] = hv;
        }
        __syncthreads();

        v16h a0 = frag_a(&lX[w * 32][0],      40, lane);
        v16h a1 = frag_a(&lX[w * 32 + 16][0], 40, lane);
#pragma unroll
        for (int c = 0; c < 4; ++c) {
            v16h b = frag_bT(&lW[c * 16][0], 40, lane);
            acc[0][c] = WMMA_F16(a0, b, acc[0][c]);
            acc[1][c] = WMMA_F16(a1, b, acc[1][c]);
        }
    }

    // Branch-free scatter-store: idx = bh*131072 + l*sL + d*sD.
    const int r_off = (lane & 16) ? 8 : 0;
    const int nl    = lane & 15;
#pragma unroll
    for (int hh = 0; hh < 2; ++hh)
#pragma unroll
        for (int c = 0; c < 4; ++c) {
            int col  = n0 + c * 16 + nl;
            int head = col >> 6, d = col & 63;
            size_t dpart = (size_t)d * sD;
#pragma unroll
            for (int i = 0; i < 8; ++i) {
                int rg = r0 + w * 32 + hh * 16 + i + r_off;   // global token row
                int b  = rg >> 11, l = rg & 2047;
                size_t idx = ((size_t)b * 16 + head) * 131072
                           + (size_t)l * sL + dpart;
                Out[idx] = (_Float16)acc[hh][c][i];
            }
        }
}

// ---------------------------------------------------------------------------
// Kernel 2: flash attention per (batch*head).  Block: 8 waves, each wave owns
// a 16-row Q strip (128 rows/block).  K (row-major) and V^T (d-major) tiles
// streamed into LDS with CDNA5 async-to-LDS copies; online softmax; P
// re-staged through LDS into the WMMA A-fragment layout.
// ---------------------------------------------------------------------------
__global__ __launch_bounds__(256) void flash_attn_kernel(
    const _Float16* __restrict__ Qh, const _Float16* __restrict__ Kh,
    const _Float16* __restrict__ Vt, _Float16* __restrict__ Oh)
{
    __shared__ _Float16 lK[64][80];      // K tile:  [j][d], 160B rows
    __shared__ _Float16 lVt[64][80];     // V^T tile:[d][j]
    __shared__ _Float16 lP[8][16][80];   // per-wave P staging

    const int tid  = threadIdx.x;
    const int lane = tid & 31;
    const int w    = tid >> 5;
    const int q0   = blockIdx.x * 128;
    const int bh   = blockIdx.y;         // b*16 + head

    const _Float16* Qb = Qh + ((size_t)bh * 2048 + q0 + w * 16) * 64;
    const _Float16* Kb = Kh + (size_t)bh * 2048 * 64;
    const _Float16* Vb = Vt + (size_t)bh * 64 * 2048;

    // Q fragments (16x64) loaded once, straight from global.
    v16h aq0 = frag_a(Qb,      64, lane);
    v16h aq1 = frag_a(Qb + 32, 64, lane);

    const v8f z8 = {0.f, 0.f, 0.f, 0.f, 0.f, 0.f, 0.f, 0.f};
    v8f acc[4] = {z8, z8, z8, z8};
    float mrow[8], lrow[8];
#pragma unroll
    for (int i = 0; i < 8; ++i) { mrow[i] = -3.0e38f; lrow[i] = 0.f; }

    const int   r_off = (lane & 16) ? 8 : 0;
    const int   nl    = lane & 15;
    const float sc_qk = 0.125f;          // 1/sqrt(64)

    // Per-thread chunks of the 64x64 tile copies (16B granules).
    const int cp_row0 = tid >> 3,         cp_off0 = (tid & 7) * 8;
    const int cp_row1 = (256 + tid) >> 3, cp_off1 = ((256 + tid) & 7) * 8;

    for (int j0 = 0; j0 < 2048; j0 += 64) {
        __syncthreads();                 // previous tile's reads finished
        // Async memory->LDS (ASYNCcnt), 4x16B per thread.
        {
            unsigned gk0 = (unsigned)(((j0 + cp_row0) * 64 + cp_off0) * 2);
            unsigned gk1 = (unsigned)(((j0 + cp_row1) * 64 + cp_off1) * 2);
            unsigned gv0 = (unsigned)((cp_row0 * 2048 + j0 + cp_off0) * 2);
            unsigned gv1 = (unsigned)((cp_row1 * 2048 + j0 + cp_off1) * 2);
            async_load_b128((unsigned)(unsigned long long)&lK [cp_row0][cp_off0], gk0, Kb);
            async_load_b128((unsigned)(unsigned long long)&lK [cp_row1][cp_off1], gk1, Kb);
            async_load_b128((unsigned)(unsigned long long)&lVt[cp_row0][cp_off0], gv0, Vb);
            async_load_b128((unsigned)(unsigned long long)&lVt[cp_row1][cp_off1], gv1, Vb);
        }
        if (j0 + 64 < 2048) {            // pull next tiles toward L2/WGP$
            __builtin_prefetch(Kb + (size_t)(j0 + 64 + (tid >> 2)) * 64, 0, 1);
            __builtin_prefetch(Vb + (size_t)(tid >> 2) * 2048 + j0 + 64, 0, 1);
        }
        wait_asynccnt0();                // this wave's LDS writes landed
        __syncthreads();                 // visible to all waves

        // S = Q K^T for this 16x64 block (4 sub-blocks of 16x16).
        v8f s[4];
#pragma unroll
        for (int jc = 0; jc < 4; ++jc) {
            v8f t = z8;
            t = WMMA_F16(aq0, frag_bT(&lK[jc * 16][0],  80, lane), t);
            t = WMMA_F16(aq1, frag_bT(&lK[jc * 16][32], 80, lane), t);
            s[jc] = t;
        }

        // Online softmax. Row r = i + r_off lives across the lane's 16-lane half.
        float rmax[8];
#pragma unroll
        for (int i = 0; i < 8; ++i) {
            float m = fmaxf(fmaxf(s[0][i], s[1][i]), fmaxf(s[2][i], s[3][i]));
            rmax[i] = m * sc_qk;
        }
#pragma unroll
        for (int d = 1; d < 16; d <<= 1)
#pragma unroll
            for (int i = 0; i < 8; ++i)
                rmax[i] = fmaxf(rmax[i], __shfl_xor(rmax[i], d, 32));

        float mnew[8], scale[8], rsum[8];
#pragma unroll
        for (int i = 0; i < 8; ++i) {
            mnew[i]  = fmaxf(mrow[i], rmax[i]);
            scale[i] = __expf(mrow[i] - mnew[i]);
            mrow[i]  = mnew[i];
            rsum[i]  = 0.f;
        }
#pragma unroll
        for (int jc = 0; jc < 4; ++jc)
#pragma unroll
            for (int i = 0; i < 8; ++i) {
                float p = __expf(s[jc][i] * sc_qk - mnew[i]);
                s[jc][i] = p;
                rsum[i] += p;
            }
#pragma unroll
        for (int d = 1; d < 16; d <<= 1)
#pragma unroll
            for (int i = 0; i < 8; ++i)
                rsum[i] += __shfl_xor(rsum[i], d, 32);
#pragma unroll
        for (int i = 0; i < 8; ++i) lrow[i] = lrow[i] * scale[i] + rsum[i];
#pragma unroll
        for (int c = 0; c < 4; ++c)
#pragma unroll
            for (int i = 0; i < 8; ++i) acc[c][i] = acc[c][i] * scale[i];

        // Stage P (16x64 f16) so it can be re-read in A-fragment layout.
#pragma unroll
        for (int jc = 0; jc < 4; ++jc)
#pragma unroll
            for (int i = 0; i < 8; ++i)
                lP[w][i + r_off][jc * 16 + nl] = (_Float16)s[jc][i];
        __syncthreads();

        // O += P @ V.  B-fragment from V^T tile ([d][j] -> [n][k]) is
        // contiguous: 2x ds_load_b128, same as the K path.
        v16h ap0 = frag_a(&lP[w][0][0],  80, lane);
        v16h ap1 = frag_a(&lP[w][0][32], 80, lane);
#pragma unroll
        for (int c = 0; c < 4; ++c) {
            acc[c] = WMMA_F16(ap0, frag_bT(&lVt[c * 16][0],  80, lane), acc[c]);
            acc[c] = WMMA_F16(ap1, frag_bT(&lVt[c * 16][32], 80, lane), acc[c]);
        }
    }

    // Normalize and store attention output as [b][l][head*64+d] f16.
    const int b = bh >> 4, head = bh & 15;
    _Float16* Ob = Oh + ((size_t)b * 2048 + q0 + w * 16) * 1024 + head * 64;
#pragma unroll
    for (int c = 0; c < 4; ++c)
#pragma unroll
        for (int i = 0; i < 8; ++i)
            Ob[(size_t)(i + r_off) * 1024 + c * 16 + nl] =
                (_Float16)(acc[c][i] / lrow[i]);
}

// ---------------------------------------------------------------------------
// Kernel 3: output projection  Y = O @ Wo^T + b_o  (f16 A, f32 W -> f32 out).
// Block: 8 waves, 256x64 output tile; 32x64 strip per wave.
// ---------------------------------------------------------------------------
__global__ __launch_bounds__(256) void out_proj_kernel(
    const _Float16* __restrict__ Oh, const float* __restrict__ Wo,
    const float* __restrict__ bo, float* __restrict__ Y)
{
    __shared__ _Float16 lA[256][40];
    __shared__ _Float16 lW[64][40];

    const int tid  = threadIdx.x;
    const int lane = tid & 31;
    const int w    = tid >> 5;
    const int r0   = blockIdx.x * 256;
    const int n0   = blockIdx.y * 64;

    const v8f z8 = {0.f, 0.f, 0.f, 0.f, 0.f, 0.f, 0.f, 0.f};
    v8f acc[2][4] = {{z8, z8, z8, z8}, {z8, z8, z8, z8}};

    for (int kk = 0; kk < 1024; kk += 32) {
        __syncthreads();
        // A tile 256x32 f16 straight copy (4x16B per thread).
#pragma unroll
        for (int p = 0; p < 4; ++p) {
            int c = p * 256 + tid;
            int row = c >> 2, off = (c & 3) * 8;
            *(float4*)&lA[row][off] = *(const float4*)(Oh + (size_t)(r0 + row) * 1024 + kk + off);
        }
        // W tile 64x32 f32 -> f16.
#pragma unroll
        for (int p = 0; p < 2; ++p) {
            int row = p * 32 + (tid >> 3);
            int col = (tid & 7) * 4;
            float4 v = *(const float4*)(Wo + (size_t)(n0 + row) * 1024 + kk + col);
            h4 hv = {(_Float16)v.x, (_Float16)v.y, (_Float16)v.z, (_Float16)v.w};
            *(h4*)&lW[row][col] = hv;
        }
        __syncthreads();

        v16h a0 = frag_a(&lA[w * 32][0],      40, lane);
        v16h a1 = frag_a(&lA[w * 32 + 16][0], 40, lane);
#pragma unroll
        for (int c = 0; c < 4; ++c) {
            v16h b = frag_bT(&lW[c * 16][0], 40, lane);
            acc[0][c] = WMMA_F16(a0, b, acc[0][c]);
            acc[1][c] = WMMA_F16(a1, b, acc[1][c]);
        }
    }

    const int r_off = (lane & 16) ? 8 : 0;
    const int nl    = lane & 15;
#pragma unroll
    for (int hh = 0; hh < 2; ++hh)
#pragma unroll
        for (int c = 0; c < 4; ++c) {
            int col = n0 + c * 16 + nl;
            float bias = bo[col];
#pragma unroll
            for (int i = 0; i < 8; ++i)
                Y[(size_t)(r0 + w * 32 + hh * 16 + i + r_off) * 1024 + col] =
                    acc[hh][c][i] + bias;
        }
}

// ---------------------------------------------------------------------------
extern "C" void kernel_launch(void* const* d_in, const int* in_sizes, int n_in,
                              void* d_out, int out_size, void* d_ws, size_t ws_size,
                              hipStream_t stream)
{
    (void)in_sizes; (void)n_in; (void)out_size; (void)ws_size;

    const float* x  = (const float*)d_in[0];
    const float* Wq = (const float*)d_in[1];
    const float* Wk = (const float*)d_in[2];
    const float* Wv = (const float*)d_in[3];
    const float* Wo = (const float*)d_in[4];
    const float* bo = (const float*)d_in[5];
    float*       y  = (float*)d_out;

    // Workspace: Q,K as [b][head][2048][64] f16; V^T as [b][head][64][2048] f16;
    // attn-out as [b][l][1024] f16.  64 MB total.
    _Float16* Qh = (_Float16*)d_ws;
    _Float16* Kh = Qh + (size_t)8192 * 1024;
    _Float16* Vt = Kh + (size_t)8192 * 1024;
    _Float16* Oh = Vt + (size_t)8192 * 1024;

    dim3 blk(256);
    qkv_proj_kernel <<<dim3(32, 16, 3), blk, 0, stream>>>(x, Wq, Wk, Wv, Qh, Kh, Vt);
    flash_attn_kernel<<<dim3(16, 64),    blk, 0, stream>>>(Qh, Kh, Vt, Oh);
    out_proj_kernel <<<dim3(32, 16),     blk, 0, stream>>>(Oh, Wo, bo, y);
}
